// DiseaseGraphAttention_35862976921707
// MI455X (gfx1250) — matrix operand order repaired and checked by reading, not compiled, not run on hardware
//
#include <hip/hip_runtime.h>

// ---------------- types ----------------
typedef __attribute__((ext_vector_type(16))) __bf16 v16bf;
typedef __attribute__((ext_vector_type(8)))  __bf16 v8bf;
typedef __attribute__((ext_vector_type(8)))  float  v8f;

#define BATCH 4
#define TLEN  1400
#define CDIM  512
#define NHEAD 8
#define HDIM  64
#define BT    (BATCH*TLEN)          // 5600
#define TPAD  1408                  // 44 * 32

// ---------------- WMMA helpers ----------------
__device__ __forceinline__ v8f wmma_bf16(v16bf a, v16bf b, v8f c) {
  return __builtin_amdgcn_wmma_f32_16x16x32_bf16(false, a, false, b, (short)0, c,
                                                 false, false);
}

// A-fragment (16x32 bf16): lane row = lane%16; 16 contiguous K at 16*(lane/16).
// rowp points at start of this lane's row.
__device__ __forceinline__ v16bf load_a16(const __bf16* rowp, int kk0, int hl) {
  const __bf16* p = rowp + kk0 + hl * 16;
  v8bf lo = *(const v8bf*)(p);
  v8bf hi = *(const v8bf*)(p + 8);
  v16bf r;
#pragma unroll
  for (int i = 0; i < 8; i++) { r[i] = lo[i]; r[i + 8] = hi[i]; }
  return r;
}

// B-fragment (32x16 bf16): lane col = lane%16; 16 contiguous K at 16*(lane/16).
// rowp points at the (transposed) row holding this lane's column.
__device__ __forceinline__ v16bf load_b16(const __bf16* rowp, int kk0, int hl) {
  return *(const v16bf*)(rowp + kk0 + hl * 16);
}

// 32x32 wave tile: 2x2 WMMA tiles, each A/B fragment reused twice.
__device__ __forceinline__ void gemm32_core(const __bf16* __restrict__ A,
                                            const __bf16* __restrict__ Bt,
                                            int m0, int n0, int l16, int hl,
                                            v8f acc[4]) {
  const __bf16* ar0 = A + (size_t)(m0 + l16) * CDIM;
  const __bf16* ar1 = A + (size_t)(m0 + 16 + l16) * CDIM;
  const __bf16* br0 = Bt + (size_t)(n0 + l16) * CDIM;
  const __bf16* br1 = Bt + (size_t)(n0 + 16 + l16) * CDIM;
#pragma unroll 2
  for (int kk0 = 0; kk0 < CDIM; kk0 += 32) {
    v16bf a0 = load_a16(ar0, kk0, hl);
    v16bf a1 = load_a16(ar1, kk0, hl);
    v16bf b0 = load_b16(br0, kk0, hl);
    v16bf b1 = load_b16(br1, kk0, hl);
    acc[0] = wmma_bf16(a0, b0, acc[0]);
    acc[1] = wmma_bf16(a0, b1, acc[1]);
    acc[2] = wmma_bf16(a1, b0, acc[2]);
    acc[3] = wmma_bf16(a1, b1, acc[3]);
  }
}

// ---------------- conversion kernels ----------------
__global__ void dga_cvt_x(const float* __restrict__ x, __bf16* __restrict__ xb, int n) {
  int i = blockIdx.x * 256 + threadIdx.x;
  if (i < n) xb[i] = (__bf16)x[i];
}

// Wt[n*512 + k] = W[k*512 + n]  (bf16 transposed weights), for 4 weight matrices
__global__ void dga_cvt_w(const float* __restrict__ Wq, const float* __restrict__ Wk,
                          const float* __restrict__ Wv, const float* __restrict__ Wo,
                          __bf16* __restrict__ wqt, __bf16* __restrict__ wkt,
                          __bf16* __restrict__ wvt, __bf16* __restrict__ wot) {
  int i = blockIdx.x * 256 + threadIdx.x;
  if (i >= 4 * CDIM * CDIM) return;
  int w   = i >> 18;
  int rem = i & (CDIM * CDIM - 1);
  int nr  = rem >> 9;
  int kc  = rem & (CDIM - 1);
  const float* W = (w == 0) ? Wq : (w == 1) ? Wk : (w == 2) ? Wv : Wo;
  __bf16* Wt     = (w == 0) ? wqt : (w == 1) ? wkt : (w == 2) ? wvt : wot;
  Wt[rem] = (__bf16)W[kc * CDIM + nr];
}

// ---------------- fused QKV projection GEMM ----------------
// q/k/v stored flat [B*T][512] (bf16): row m = b*T + t, col n = h*64 + d.
// grid (175, 4, 3), block 128; wave w -> 32x32 tile at (blockIdx.x*32, (blockIdx.y*4+w)*32)
__global__ __launch_bounds__(128) void dga_gemm_qkv(
    const __bf16* __restrict__ xb,
    const __bf16* __restrict__ wqt, const __bf16* __restrict__ wkt,
    const __bf16* __restrict__ wvt,
    const float* __restrict__ bq, const float* __restrict__ bk,
    const float* __restrict__ bv,
    __bf16* __restrict__ qo, __bf16* __restrict__ ko, __bf16* __restrict__ vo) {
  int wave = threadIdx.x >> 5, lane = threadIdx.x & 31;
  int l16 = lane & 15, hl = lane >> 4;
  int m0 = blockIdx.x * 32;
  int n0 = (blockIdx.y * 4 + wave) * 32;
  int z = blockIdx.z;
  const __bf16* wt = (z == 0) ? wqt : (z == 1) ? wkt : wvt;
  const float* bias = (z == 0) ? bq : (z == 1) ? bk : bv;
  __bf16* out = (z == 0) ? qo : (z == 1) ? ko : vo;

  v8f zero = {};
  v8f acc[4] = {zero, zero, zero, zero};
  gemm32_core(xb, wt, m0, n0, l16, hl, acc);

#pragma unroll
  for (int mi = 0; mi < 2; mi++) {
#pragma unroll
    for (int ni = 0; ni < 2; ni++) {
      int n = n0 + ni * 16 + l16;
      float bn = bias[n];
#pragma unroll
      for (int r = 0; r < 8; r++) {
        int m = m0 + mi * 16 + r + 8 * hl;
        out[(size_t)m * CDIM + n] = (__bf16)(acc[mi * 2 + ni][r] + bn);
      }
    }
  }
}

// ---------------- fused flash attention ----------------
// q/k/v flat [B*T][512]; per (b,h) use row-stride 512, column offset h*64.
// grid: (22, H, B), block 128 (4 waves); wave w handles query tile blockIdx.x*4+w
__global__ __launch_bounds__(128) void dga_attn(
    const __bf16* __restrict__ q, const __bf16* __restrict__ k,
    const __bf16* __restrict__ v, const int* __restrict__ ids,
    const float* __restrict__ aff, __bf16* __restrict__ ao) {
  __shared__ __align__(32) __bf16 Klds[32 * 64];        // [key][d]
  __shared__ __align__(32) __bf16 Vtlds[64 * 32];       // [d][key]
  __shared__ __align__(32) __bf16 Plds[4][16 * 32];     // per-wave P tile

  int tid = threadIdx.x;
  int wave = tid >> 5, lane = tid & 31;
  int l16 = lane & 15, hl = lane >> 4;
  int bi = blockIdx.z, h = blockIdx.y;
  int q0 = (blockIdx.x * 4 + wave) * 16;

  const __bf16* qb = q + (size_t)bi * TLEN * CDIM + h * HDIM;  // row stride CDIM
  const __bf16* kb = k + (size_t)bi * TLEN * CDIM + h * HDIM;
  const __bf16* vb = v + (size_t)bi * TLEN * CDIM + h * HDIM;

  // Q fragments for d = 0..31 and 32..63, kept live across the key loop
  int tq = q0 + l16; if (tq > TLEN - 1) tq = TLEN - 1;
  v16bf qf0 = load_a16(qb + (size_t)tq * CDIM, 0, hl);
  v16bf qf1 = load_a16(qb + (size_t)tq * CDIM, 32, hl);

  int idr[8];
#pragma unroll
  for (int r = 0; r < 8; r++) {
    int t = q0 + r + 8 * hl; if (t > TLEN - 1) t = TLEN - 1;
    idr[r] = ids[bi * TLEN + t];
  }

  // cooperative-loader mapping: 4 threads per key row, 16 contiguous d each
  int ckey = tid >> 2;          // 0..31
  int cd0  = (tid & 3) * 16;    // 0,16,32,48

  float mrow[8], lrow[8];
  v8f zero = {};
  v8f oacc[4];
#pragma unroll
  for (int r = 0; r < 8; r++) { mrow[r] = -3.0e38f; lrow[r] = 0.0f; }
#pragma unroll
  for (int nt = 0; nt < 4; nt++) oacc[nt] = zero;

  for (int kt0 = 0; kt0 < TPAD; kt0 += 32) {
    __syncthreads();  // protect previous iteration's LDS reads
    // cooperative load: K tile (32x64) row-major via b128s, V transposed scatter
    {
      int ks = kt0 + ckey; if (ks > TLEN - 1) ks = TLEN - 1;
      const __bf16* krow = kb + (size_t)ks * CDIM + cd0;
      const __bf16* vrow = vb + (size_t)ks * CDIM + cd0;
      v8bf k0 = *(const v8bf*)(krow);
      v8bf k1 = *(const v8bf*)(krow + 8);
      v8bf w0 = *(const v8bf*)(vrow);
      v8bf w1 = *(const v8bf*)(vrow + 8);
      *(v8bf*)&Klds[ckey * 64 + cd0]     = k0;
      *(v8bf*)&Klds[ckey * 64 + cd0 + 8] = k1;
#pragma unroll
      for (int i = 0; i < 8; i++) {
        Vtlds[(cd0 + i) * 32 + ckey]     = w0[i];
        Vtlds[(cd0 + 8 + i) * 32 + ckey] = w1[i];
      }
    }
    __syncthreads();

    // scores: two 16x16 tiles (keys kt0..+15 and kt0+16..+31), contraction D=64
    v8f s0 = zero, s1 = zero;
    s0 = wmma_bf16(qf0, load_b16(&Klds[l16 * 64], 0, hl), s0);
    s0 = wmma_bf16(qf1, load_b16(&Klds[l16 * 64], 32, hl), s0);
    s1 = wmma_bf16(qf0, load_b16(&Klds[(16 + l16) * 64], 0, hl), s1);
    s1 = wmma_bf16(qf1, load_b16(&Klds[(16 + l16) * 64], 32, hl), s1);

    int c0 = kt0 + l16, c1 = kt0 + 16 + l16;
    int cc0 = (c0 < TLEN) ? c0 : 0;
    int cc1 = (c1 < TLEN) ? c1 : 0;

#pragma unroll
    for (int r = 0; r < 8; r++) {
      float s0v = s0[r] * 0.125f + 0.1f * aff[(size_t)idr[r] * TLEN + cc0];
      float s1v = s1[r] * 0.125f + 0.1f * aff[(size_t)idr[r] * TLEN + cc1];
      if (c0 >= TLEN) s0v = -3.0e38f;
      if (c1 >= TLEN) s1v = -3.0e38f;
      // row max over the 16 lanes of this half (xor 1,2,4,8 stays in-half)
      float rm = fmaxf(s0v, s1v);
#pragma unroll
      for (int off = 1; off < 16; off <<= 1) rm = fmaxf(rm, __shfl_xor(rm, off, 32));
      float mn = fmaxf(mrow[r], rm);
      float corr = __expf(mrow[r] - mn);
      float p0 = __expf(s0v - mn);
      float p1 = __expf(s1v - mn);
      float ps = p0 + p1;
#pragma unroll
      for (int off = 1; off < 16; off <<= 1) ps += __shfl_xor(ps, off, 32);
      lrow[r] = lrow[r] * corr + ps;
      mrow[r] = mn;
#pragma unroll
      for (int nt = 0; nt < 4; nt++) oacc[nt][r] *= corr;
      int prow = (r + 8 * hl) * 32;
      Plds[wave][prow + l16]      = (__bf16)p0;
      Plds[wave][prow + 16 + l16] = (__bf16)p1;
    }

    // O += P (16x32) @ V (32x64), split over 4 d-chunks of 16
    v16bf pa = load_a16(&Plds[wave][l16 * 32], 0, hl);
#pragma unroll
    for (int nt = 0; nt < 4; nt++) {
      v16bf bv_ = load_b16(&Vtlds[(nt * 16 + l16) * 32], 0, hl);
      oacc[nt] = wmma_bf16(pa, bv_, oacc[nt]);
    }
  }

  // epilogue: normalize and store to ao[b*T + t][h*64 + d] (bf16)
#pragma unroll
  for (int r = 0; r < 8; r++) {
    float inv = (lrow[r] > 0.0f) ? 1.0f / lrow[r] : 0.0f;
    int t = q0 + r + 8 * hl;
    if (t < TLEN) {
#pragma unroll
      for (int nt = 0; nt < 4; nt++) {
        ao[((size_t)(bi * TLEN + t)) * CDIM + h * HDIM + nt * 16 + l16] =
            (__bf16)(oacc[nt][r] * inv);
      }
    }
  }
}

// ---------------- output projection GEMM (f32 out) ----------------
// grid (175, 4), block 128; 32x32 tile per wave
__global__ __launch_bounds__(128) void dga_gemm_out(
    const __bf16* __restrict__ ao, const __bf16* __restrict__ wot,
    const float* __restrict__ bo, float* __restrict__ out) {
  int wave = threadIdx.x >> 5, lane = threadIdx.x & 31;
  int l16 = lane & 15, hl = lane >> 4;
  int m0 = blockIdx.x * 32;
  int n0 = (blockIdx.y * 4 + wave) * 32;

  v8f zero = {};
  v8f acc[4] = {zero, zero, zero, zero};
  gemm32_core(ao, wot, m0, n0, l16, hl, acc);

#pragma unroll
  for (int mi = 0; mi < 2; mi++) {
#pragma unroll
    for (int ni = 0; ni < 2; ni++) {
      int n = n0 + ni * 16 + l16;
      float bn = bo[n];
#pragma unroll
      for (int r = 0; r < 8; r++) {
        int m = m0 + mi * 16 + r + 8 * hl;
        out[(size_t)m * CDIM + n] = acc[mi * 2 + ni][r] + bn;
      }
    }
  }
}

// ---------------- host launcher ----------------
extern "C" void kernel_launch(void* const* d_in, const int* in_sizes, int n_in,
                              void* d_out, int out_size, void* d_ws, size_t ws_size,
                              hipStream_t stream) {
  const float* x   = (const float*)d_in[0];
  const int*   ids = (const int*)d_in[1];
  const float* Wq  = (const float*)d_in[2];
  const float* bq  = (const float*)d_in[3];
  const float* Wk  = (const float*)d_in[4];
  const float* bk  = (const float*)d_in[5];
  const float* Wv  = (const float*)d_in[6];
  const float* bv  = (const float*)d_in[7];
  const float* aff = (const float*)d_in[8];
  const float* Wo  = (const float*)d_in[9];
  const float* bo  = (const float*)d_in[10];
  float* out = (float*)d_out;

  char* ws = (char*)d_ws;
  size_t off = 0;
  auto alloc = [&](size_t bytes) -> char* {
    char* p = ws + off;
    off += (bytes + 255) & ~(size_t)255;
    return p;
  };
  const size_t xbytes = (size_t)BT * CDIM * 2;          // 5,734,400
  const size_t wbytes = (size_t)CDIM * CDIM * 2;        // 524,288
  __bf16* xb  = (__bf16*)alloc(xbytes);
  __bf16* wqt = (__bf16*)alloc(wbytes);
  __bf16* wkt = (__bf16*)alloc(wbytes);
  __bf16* wvt = (__bf16*)alloc(wbytes);
  __bf16* wot = (__bf16*)alloc(wbytes);
  __bf16* qd  = (__bf16*)alloc(xbytes);
  __bf16* kd  = (__bf16*)alloc(xbytes);
  __bf16* vd  = (__bf16*)alloc(xbytes);
  __bf16* aod = (__bf16*)alloc(xbytes);

  int nx = BT * CDIM;
  dga_cvt_x<<<(nx + 255) / 256, 256, 0, stream>>>(x, xb, nx);
  dga_cvt_w<<<(4 * CDIM * CDIM + 255) / 256, 256, 0, stream>>>(
      Wq, Wk, Wv, Wo, wqt, wkt, wvt, wot);

  dim3 g1(BT / 32, CDIM / 128, 3);  // (175, 4, 3)
  dga_gemm_qkv<<<g1, 128, 0, stream>>>(xb, wqt, wkt, wvt, bq, bk, bv, qd, kd, vd);

  dim3 g2(TPAD / 64, NHEAD, BATCH); // (22, 8, 4)
  dga_attn<<<g2, 128, 0, stream>>>(qd, kd, vd, ids, aff, aod);

  dim3 g3(BT / 32, CDIM / 128, 1);  // (175, 4, 1)
  dga_gemm_out<<<g3, 128, 0, stream>>>(aod, wot, bo, out);
}
